// GATModelSimple_75213467288113
// MI455X (gfx1250) — matrix operand order repaired and checked by reading, not compiled, not run on hardware
//
#include <hip/hip_runtime.h>

// ---------------------------------------------------------------------------
// GAT (2-layer) for MI455X / gfx1250.
// Layer-1 feature transform runs on v_wmma_f32_16x16x32_bf16 (f32 accum).
// Edge softmax/aggregation phases are HBM-bound scatter with f32 atomics.
// ---------------------------------------------------------------------------

typedef __attribute__((ext_vector_type(16))) __bf16 v16bf;
typedef __attribute__((ext_vector_type(8)))  __bf16 v8bf;
typedef __attribute__((ext_vector_type(8)))  float  v8f;

#define F_IN      166
#define KPAD      192           // F_IN padded to multiple of 32 (K tile)
#define HCH       256           // HEADS * HID
#define HEADS     4
#define HID       64
#define NEG_SLOPE 0.2f

static __device__ __forceinline__ unsigned short f2bf(float f) {
  // round-to-nearest-even f32 -> bf16
  unsigned int u = __float_as_uint(f);
  unsigned int r = ((u >> 16) & 1u) + 0x7FFFu;
  return (unsigned short)((u + r) >> 16);
}

static __device__ __forceinline__ float lrelu(float x) {
  return x > 0.f ? x : NEG_SLOPE * x;
}

static __device__ __forceinline__ float neg_inf() {
  return __uint_as_float(0xFF800000u);
}

// float atomic max via int/uint atomics (valid with -inf init)
static __device__ __forceinline__ void atomicMaxF(float* addr, float val) {
  if (val >= 0.f) atomicMax((int*)addr, __float_as_int(val));
  else            atomicMin((unsigned int*)addr, __float_as_uint(val));
}

// ---------------------------------------------------------------------------
// fills / conversions
// ---------------------------------------------------------------------------
__global__ void fill_f32(float* __restrict__ p, float v, long long n) {
  long long i = (long long)blockIdx.x * blockDim.x + threadIdx.x;
  if (i < n) p[i] = v;
}

// x [N, F_IN] f32 -> xb [N, KPAD] bf16 (zero padded)
__global__ void cvt_x_kernel(const float* __restrict__ x,
                             unsigned short* __restrict__ xb, int nNodes) {
  long long i = (long long)blockIdx.x * blockDim.x + threadIdx.x;
  long long total = (long long)nNodes * KPAD;
  if (i >= total) return;
  int k = (int)(i % KPAD);
  long long row = i / KPAD;
  xb[i] = (k < F_IN) ? f2bf(x[row * F_IN + k]) : (unsigned short)0;
}

// W1 [F_IN, HCH] f32 -> wt [HCH, KPAD] bf16 (transposed, zero padded)
__global__ void cvt_w1_kernel(const float* __restrict__ w,
                              unsigned short* __restrict__ wt) {
  int i = blockIdx.x * blockDim.x + threadIdx.x;
  if (i >= HCH * KPAD) return;
  int k = i % KPAD;
  int n = i / KPAD;
  wt[i] = (k < F_IN) ? f2bf(w[k * HCH + n]) : (unsigned short)0;
}

// ---------------------------------------------------------------------------
// GEMM1: h1[N,256] = x[N,166] @ W1[166,256]  via v_wmma_f32_16x16x32_bf16
// One wave -> 16-row x 128-col strip (8 WMMA tiles, 6 K-steps = 48 WMMA).
// ---------------------------------------------------------------------------
__global__ __launch_bounds__(256)
void gemm1_wmma(const unsigned short* __restrict__ xb_,
                const unsigned short* __restrict__ wt_,
                float* __restrict__ h1, int nNodes) {
  const __bf16* __restrict__ xb = (const __bf16*)xb_;
  const __bf16* __restrict__ wt = (const __bf16*)wt_;

  const int lane  = threadIdx.x & 31;
  const int wave  = threadIdx.x >> 5;
  const int gw    = blockIdx.x * 8 + wave;
  const int strip = gw >> 1;                    // 16-row strip index
  const int nBase = (gw & 1) * 128;             // which 128-col half
  const int mBase = strip * 16;
  if (mBase >= nNodes) return;                  // uniform per-wave exit

  const int r  = lane & 15;
  const int hi = lane >> 4;                     // 0 or 1

  v8f acc[8];
#pragma unroll
  for (int i = 0; i < 8; ++i) acc[i] = (v8f){};

  int row = mBase + r;
  if (row >= nNodes) row = nNodes - 1;          // clamp loads (stores guarded)
  const __bf16* aRow = xb + (size_t)row * KPAD;

#pragma unroll
  for (int kt = 0; kt < 6; ++kt) {
    // A fragment: 16x32 bf16, k = 16*hi + s (contiguous, same as B)
    const int kb = kt * 32 + hi * 16;
    union { v16bf v; v8bf h[2]; } a;
    a.h[0] = *(const v8bf*)(aRow + kb);
    a.h[1] = *(const v8bf*)(aRow + kb + 8);

    const int kbB = kt * 32 + hi * 16;
#pragma unroll
    for (int nt = 0; nt < 8; ++nt) {
      const int col = nBase + nt * 16 + r;
      v16bf b = *(const v16bf*)(wt + (size_t)col * KPAD + kbB);
      acc[nt] = __builtin_amdgcn_wmma_f32_16x16x32_bf16(
          false, a.v, false, b, (short)0, acc[nt], false, false);
    }
  }

  // C/D layout: VGPR rr -> M = mBase + rr + hi*8, N = nBase + nt*16 + r
#pragma unroll
  for (int nt = 0; nt < 8; ++nt) {
    const int col = nBase + nt * 16 + r;
#pragma unroll
    for (int rr = 0; rr < 8; ++rr) {
      const int m = mBase + rr + hi * 8;
      if (m < nNodes) h1[(size_t)m * HCH + col] = acc[nt][rr];
    }
  }
}

// ---------------------------------------------------------------------------
// Layer-1 attention logits per (node, head); also init m1/-inf, dn1/0
// ---------------------------------------------------------------------------
__global__ void alpha1_kernel(const float* __restrict__ h1,
                              const float* __restrict__ aS,
                              const float* __restrict__ aD,
                              float* __restrict__ as1, float* __restrict__ ad1,
                              float* __restrict__ m1, float* __restrict__ dn1,
                              int nNodes) {
  int idx = blockIdx.x * blockDim.x + threadIdx.x;
  if (idx >= nNodes * HEADS) return;
  int n = idx >> 2, hd = idx & 3;
  const float* r  = h1 + (size_t)n * HCH + hd * HID;
  const float* ws = aS + hd * HID;
  const float* wd = aD + hd * HID;
  float s = 0.f, d = 0.f;
#pragma unroll 8
  for (int c = 0; c < HID; ++c) { float v = r[c]; s += v * ws[c]; d += v * wd[c]; }
  as1[idx] = s; ad1[idx] = d;
  m1[idx]  = neg_inf(); dn1[idx] = 0.f;
}

// ---------------------------------------------------------------------------
// Edge passes, layer 1 (E real edges + N self loops)
// ---------------------------------------------------------------------------
__global__ void edge_max1(const int* __restrict__ ei, long long E, int nNodes,
                          const float* __restrict__ as1,
                          const float* __restrict__ ad1,
                          float* __restrict__ m1) {
  long long e = (long long)blockIdx.x * blockDim.x + threadIdx.x;
  if (e >= E + nNodes) return;
  int s, d;
  if (e < E) { s = ei[e]; d = ei[E + e]; } else { s = d = (int)(e - E); }
  float4 sv = ((const float4*)as1)[s];
  float4 dv = ((const float4*)ad1)[d];
  atomicMaxF(&m1[4 * d + 0], lrelu(sv.x + dv.x));
  atomicMaxF(&m1[4 * d + 1], lrelu(sv.y + dv.y));
  atomicMaxF(&m1[4 * d + 2], lrelu(sv.z + dv.z));
  atomicMaxF(&m1[4 * d + 3], lrelu(sv.w + dv.w));
}

__global__ void edge_sum1(const int* __restrict__ ei, long long E, int nNodes,
                          const float* __restrict__ as1,
                          const float* __restrict__ ad1,
                          const float* __restrict__ m1,
                          float* __restrict__ dn1) {
  long long e = (long long)blockIdx.x * blockDim.x + threadIdx.x;
  if (e >= E + nNodes) return;
  int s, d;
  if (e < E) { s = ei[e]; d = ei[E + e]; } else { s = d = (int)(e - E); }
  float4 sv = ((const float4*)as1)[s];
  float4 dv = ((const float4*)ad1)[d];
  float4 mv = ((const float4*)m1)[d];
  atomicAdd(&dn1[4 * d + 0], expf(lrelu(sv.x + dv.x) - mv.x));
  atomicAdd(&dn1[4 * d + 1], expf(lrelu(sv.y + dv.y) - mv.y));
  atomicAdd(&dn1[4 * d + 2], expf(lrelu(sv.z + dv.z) - mv.z));
  atomicAdd(&dn1[4 * d + 3], expf(lrelu(sv.w + dv.w) - mv.w));
}

// one wave per edge; lane handles 8 contiguous channels (head = lane>>3)
__global__ __launch_bounds__(256)
void edge_agg1(const int* __restrict__ ei, long long E, int nNodes,
               const float* __restrict__ h1,
               const float* __restrict__ as1, const float* __restrict__ ad1,
               const float* __restrict__ m1,  const float* __restrict__ dn1,
               float* __restrict__ out1) {
  long long e = (long long)blockIdx.x * 8 + (threadIdx.x >> 5);
  if (e >= E + nNodes) return;
  const int lane = threadIdx.x & 31;
  int s, d;
  if (e < E) { s = ei[e]; d = ei[E + e]; } else { s = d = (int)(e - E); }
  const int hd = lane >> 3;                       // head for my 8 channels
  float sva = as1[4 * s + hd];
  float dva = ad1[4 * d + hd];
  float mva = m1[4 * d + hd];
  float nva = dn1[4 * d + hd];
  float aa = expf(lrelu(sva + dva) - mva) / (nva + 1e-16f);

  const int cbase = lane * 8;
  const float4* hs = (const float4*)(h1 + (size_t)s * HCH + cbase);
  float4 v0 = hs[0], v1 = hs[1];
  float* od = out1 + (size_t)d * HCH + cbase;
  atomicAdd(od + 0, v0.x * aa); atomicAdd(od + 1, v0.y * aa);
  atomicAdd(od + 2, v0.z * aa); atomicAdd(od + 3, v0.w * aa);
  atomicAdd(od + 4, v1.x * aa); atomicAdd(od + 5, v1.y * aa);
  atomicAdd(od + 6, v1.z * aa); atomicAdd(od + 7, v1.w * aa);
}

// out1 = elu(out1 + b1)
__global__ void finalize1(float* __restrict__ out1, const float* __restrict__ b1,
                          long long n) {
  long long i = (long long)blockIdx.x * blockDim.x + threadIdx.x;
  if (i >= n) return;
  float v = out1[i] + b1[i & (HCH - 1)];
  out1[i] = v > 0.f ? v : (expf(v) - 1.f);
}

// ---------------------------------------------------------------------------
// Layer 2 (256 -> 2, 1 head): per-node transform + logits; init m2/dn2/out
// ---------------------------------------------------------------------------
__global__ void l2_node(const float* __restrict__ out1, const float* __restrict__ W2,
                        const float* __restrict__ aS2, const float* __restrict__ aD2,
                        float* __restrict__ h2, float* __restrict__ as2,
                        float* __restrict__ ad2, float* __restrict__ m2,
                        float* __restrict__ dn2, float* __restrict__ out,
                        int nNodes) {
  int n = blockIdx.x * blockDim.x + threadIdx.x;
  if (n >= nNodes) return;
  const float* r = out1 + (size_t)n * HCH;
  float a0 = 0.f, a1 = 0.f;
#pragma unroll 8
  for (int k = 0; k < HCH; ++k) { float v = r[k]; a0 += v * W2[2 * k]; a1 += v * W2[2 * k + 1]; }
  h2[2 * n] = a0; h2[2 * n + 1] = a1;
  as2[n] = a0 * aS2[0] + a1 * aS2[1];
  ad2[n] = a0 * aD2[0] + a1 * aD2[1];
  m2[n] = neg_inf(); dn2[n] = 0.f;
  out[2 * n] = 0.f; out[2 * n + 1] = 0.f;
}

__global__ void edge_max2(const int* __restrict__ ei, long long E, int nNodes,
                          const float* __restrict__ as2, const float* __restrict__ ad2,
                          float* __restrict__ m2) {
  long long e = (long long)blockIdx.x * blockDim.x + threadIdx.x;
  if (e >= E + nNodes) return;
  int s, d;
  if (e < E) { s = ei[e]; d = ei[E + e]; } else { s = d = (int)(e - E); }
  atomicMaxF(&m2[d], lrelu(as2[s] + ad2[d]));
}

__global__ void edge_sum2(const int* __restrict__ ei, long long E, int nNodes,
                          const float* __restrict__ as2, const float* __restrict__ ad2,
                          const float* __restrict__ m2, float* __restrict__ dn2) {
  long long e = (long long)blockIdx.x * blockDim.x + threadIdx.x;
  if (e >= E + nNodes) return;
  int s, d;
  if (e < E) { s = ei[e]; d = ei[E + e]; } else { s = d = (int)(e - E); }
  atomicAdd(&dn2[d], expf(lrelu(as2[s] + ad2[d]) - m2[d]));
}

__global__ void edge_agg2(const int* __restrict__ ei, long long E, int nNodes,
                          const float* __restrict__ as2, const float* __restrict__ ad2,
                          const float* __restrict__ m2, const float* __restrict__ dn2,
                          const float* __restrict__ h2, float* __restrict__ out) {
  long long e = (long long)blockIdx.x * blockDim.x + threadIdx.x;
  if (e >= E + nNodes) return;
  int s, d;
  if (e < E) { s = ei[e]; d = ei[E + e]; } else { s = d = (int)(e - E); }
  float al = expf(lrelu(as2[s] + ad2[d]) - m2[d]) / (dn2[d] + 1e-16f);
  atomicAdd(&out[2 * d + 0], h2[2 * s + 0] * al);
  atomicAdd(&out[2 * d + 1], h2[2 * s + 1] * al);
}

__global__ void finalize2(float* __restrict__ out, const float* __restrict__ b2,
                          int n2) {
  int i = blockIdx.x * blockDim.x + threadIdx.x;
  if (i >= n2) return;
  out[i] += b2[i & 1];
}

// ---------------------------------------------------------------------------
extern "C" void kernel_launch(void* const* d_in, const int* in_sizes, int n_in,
                              void* d_out, int out_size, void* d_ws, size_t ws_size,
                              hipStream_t stream) {
  (void)n_in; (void)out_size; (void)ws_size;

  const float* x   = (const float*)d_in[0];
  const int*   ei  = (const int*)d_in[1];
  const float* W1  = (const float*)d_in[2];
  const float* aS1 = (const float*)d_in[3];
  const float* aD1 = (const float*)d_in[4];
  const float* b1  = (const float*)d_in[5];
  const float* W2  = (const float*)d_in[6];
  const float* aS2 = (const float*)d_in[7];
  const float* aD2 = (const float*)d_in[8];
  const float* b2  = (const float*)d_in[9];
  float* out = (float*)d_out;

  const int       nNodes = in_sizes[0] / F_IN;
  const long long E      = (long long)in_sizes[1] / 2;
  const long long Etot   = E + nNodes;

  // ---- carve workspace ----
  char*  ws  = (char*)d_ws;
  size_t off = 0;
  auto carve = [&](size_t bytes) -> void* {
    void* p = ws + off;
    off += (bytes + 255) & ~(size_t)255;
    return p;
  };
  unsigned short* xb  = (unsigned short*)carve((size_t)nNodes * KPAD * 2);
  unsigned short* wt  = (unsigned short*)carve((size_t)HCH * KPAD * 2);
  float* h1   = (float*)carve((size_t)nNodes * HCH * 4);
  float* out1 = (float*)carve((size_t)nNodes * HCH * 4);
  float* as1  = (float*)carve((size_t)nNodes * HEADS * 4);
  float* ad1  = (float*)carve((size_t)nNodes * HEADS * 4);
  float* m1   = (float*)carve((size_t)nNodes * HEADS * 4);
  float* dn1  = (float*)carve((size_t)nNodes * HEADS * 4);
  float* h2   = (float*)carve((size_t)nNodes * 2 * 4);
  float* as2  = (float*)carve((size_t)nNodes * 4);
  float* ad2  = (float*)carve((size_t)nNodes * 4);
  float* m2   = (float*)carve((size_t)nNodes * 4);
  float* dn2  = (float*)carve((size_t)nNodes * 4);

  const int T = 256;
  auto blocks = [](long long n, int t) { return (int)((n + t - 1) / t); };

  // 1) bf16 conversions
  cvt_x_kernel<<<blocks((long long)nNodes * KPAD, T), T, 0, stream>>>(x, xb, nNodes);
  cvt_w1_kernel<<<blocks((long long)HCH * KPAD, T), T, 0, stream>>>(W1, wt);

  // 2) GEMM1 on WMMA: 2 waves per 16-row strip, 8 waves per block
  {
    int strips = (nNodes + 15) / 16;
    int waves  = strips * 2;
    gemm1_wmma<<<(waves + 7) / 8, T, 0, stream>>>(xb, wt, h1, nNodes);
  }

  // 3) logits + init; zero accumulator
  alpha1_kernel<<<blocks((long long)nNodes * HEADS, T), T, 0, stream>>>(
      h1, aS1, aD1, as1, ad1, m1, dn1, nNodes);
  fill_f32<<<blocks((long long)nNodes * HCH, T), T, 0, stream>>>(
      out1, 0.f, (long long)nNodes * HCH);

  // 4) edge softmax + aggregate, layer 1
  edge_max1<<<blocks(Etot, T), T, 0, stream>>>(ei, E, nNodes, as1, ad1, m1);
  edge_sum1<<<blocks(Etot, T), T, 0, stream>>>(ei, E, nNodes, as1, ad1, m1, dn1);
  edge_agg1<<<blocks(Etot * 32, T), T, 0, stream>>>(
      ei, E, nNodes, h1, as1, ad1, m1, dn1, out1);
  finalize1<<<blocks((long long)nNodes * HCH, T), T, 0, stream>>>(
      out1, b1, (long long)nNodes * HCH);

  // 5) layer 2
  l2_node<<<blocks(nNodes, T), T, 0, stream>>>(
      out1, W2, aS2, aD2, h2, as2, ad2, m2, dn2, out, nNodes);
  edge_max2<<<blocks(Etot, T), T, 0, stream>>>(ei, E, nNodes, as2, ad2, m2);
  edge_sum2<<<blocks(Etot, T), T, 0, stream>>>(ei, E, nNodes, as2, ad2, m2, dn2);
  edge_agg2<<<blocks(Etot, T), T, 0, stream>>>(ei, E, nNodes, as2, ad2, m2, dn2, h2, out);
  finalize2<<<blocks((long long)nNodes * 2, T), T, 0, stream>>>(out, b2, nNodes * 2);
}
